// FastSpeech_88235808129084
// MI455X (gfx1250) — matrix operand order repaired and checked by reading, not compiled, not run on hardware
//
#include <hip/hip_runtime.h>
#include <cstdint>

#define EMB 384
#define HID 1536
#define NHEADS 2
#define HD 192
#define NLAYERS 6
#define MELS 80
#define BB 32
#define LL 128
#define TT 1024

typedef _Float16 half_t;
typedef __attribute__((ext_vector_type(16))) _Float16 v16h;
typedef __attribute__((ext_vector_type(8)))  float    v8f;

union Frag { v16h v; uint4 q[2]; };

__device__ __forceinline__ v8f wmma16(v16h a, v16h b, v8f c) {
  return __builtin_amdgcn_wmma_f32_16x16x32_f16(false, a, false, b, (short)0, c, false, false);
}

// CDNA5 async memory->LDS copy (ISA 15.18.3 op 98, ASYNCcnt).
// lds_off: byte offset into the wave's LDS allocation (dynamic smem starts at 0).
__device__ __forceinline__ void async_load_b128(uint32_t lds_off, const void* gaddr) {
  asm volatile("global_load_async_to_lds_b128 %0, %1, off"
               :: "v"(lds_off), "v"((uint64_t)(uintptr_t)gaddr) : "memory");
}
__device__ __forceinline__ void wait_async0() {
  asm volatile("s_wait_asynccnt 0x0" ::: "memory");
}

// A-fragment (16x32 f16): lane ln = row M, halves j -> k = 16*(j>>3) + 8*hi + (j&7)
__device__ __forceinline__ v16h frag_a(const half_t* tile, int ld, int row0, int ln, int hi) {
  Frag f;
  const half_t* p = tile + (size_t)(row0 + ln) * ld + 8 * hi;
  f.q[0] = *(const uint4*)p;
  f.q[1] = *(const uint4*)(p + 16);
  return f.v;
}
// B-fragment (32x16 f16): lane ln = col N, halves j -> k = 16*(j>>3) + 8*hi + (j&7)
__device__ __forceinline__ v16h frag_b(const half_t* tile, int ld, int col0, int ln, int hi) {
  Frag f;
  const half_t* p = tile + (size_t)(col0 + ln) * ld + 8 * hi;
  f.q[0] = *(const uint4*)p;
  f.q[1] = *(const uint4*)(p + 16);
  return f.v;
}

// ---------------------------------------------------------------------------
// Unified GEMM / 1D-conv (k taps, zero pad) kernel.
// Y[M,N] = sum_t X_shifted(t)[M,K] * W[t][N,K]^T + bias, optional ReLU.
// Plain GEMM: ntaps=1, pad=0, S=M. Conv: per-batch sequences of length S.
// Block = 128 threads (4 waves, 2x2 wave grid), C-tile 64x64, K-step 32.
// Interior blocks stage tiles with global_load_async_to_lds_b128.
// Dynamic LDS: As = 64x40 halfs @0, Bs = 64x40 halfs @5120B.
// ---------------------------------------------------------------------------
#define GEMM_LDS_BYTES (2 * 64 * 40 * 2)
#define BS_OFF_BYTES   (64 * 40 * 2)

__global__ void __launch_bounds__(128) gemm_conv_kernel(
    const half_t* __restrict__ X, const half_t* __restrict__ W,
    const float* __restrict__ bias, half_t* __restrict__ Yh, float* __restrict__ Yf,
    int M, int N, int K, int S, int ntaps, int pad, int relu)
{
  extern __shared__ half_t smem[];
  half_t* As = smem;                 // 64*40
  half_t* Bs = smem + 64 * 40;       // 64*40
  const int tid = threadIdx.x;
  const int lane = tid & 31, w = tid >> 5;
  const int hi = lane >> 4, ln = lane & 15;
  const int wm = w >> 1, wn = w & 1;
  const int m0 = blockIdx.y * 64, n0 = blockIdx.x * 64;

  const int b0 = m0 / S;
  const int s0 = m0 - b0 * S;
  const bool mnFast = (m0 + 64 <= M) && (n0 + 64 <= N) && ((m0 + 63) / S == b0);

  v8f z = {0.f,0.f,0.f,0.f,0.f,0.f,0.f,0.f};
  v8f acc[2][2];
  acc[0][0] = z; acc[0][1] = z; acc[1][0] = z; acc[1][1] = z;

  for (int t = 0; t < ntaps; ++t) {
    const int shift = t - pad;
    const bool fast = mnFast && (s0 + shift >= 0) && (s0 + 63 + shift < S);
    for (int k0 = 0; k0 < K; k0 += 32) {
      if (fast) {
        #pragma unroll
        for (int ci = 0; ci < 2; ++ci) {
          int c = tid + ci * 128;              // 256 chunks of 8 halfs
          int row = c >> 2, kk0 = (c & 3) << 3;
          uint32_t lo = (uint32_t)((row * 40 + kk0) * 2);
          async_load_b128(lo,
              X + (size_t)(m0 + row + shift) * K + k0 + kk0);
          async_load_b128(lo + BS_OFF_BYTES,
              W + ((size_t)t * N + n0 + row) * K + k0 + kk0);
        }
        wait_async0();
      } else {
        #pragma unroll
        for (int ci = 0; ci < 2; ++ci) {
          int c = tid + ci * 128;
          int row = c >> 2, kk0 = (c & 3) << 3;
          uint4 va = {0u,0u,0u,0u};
          int gr = m0 + row;
          if (gr < M) {
            int bb = gr / S;
            int ss = gr - bb * S + shift;
            if (ss >= 0 && ss < S)
              va = *(const uint4*)(X + ((size_t)bb * S + ss) * K + k0 + kk0);
          }
          *(uint4*)(As + row * 40 + kk0) = va;
          uint4 vb = {0u,0u,0u,0u};
          int gn = n0 + row;
          if (gn < N)
            vb = *(const uint4*)(W + ((size_t)t * N + gn) * K + k0 + kk0);
          *(uint4*)(Bs + row * 40 + kk0) = vb;
        }
      }
      __syncthreads();
      v16h a0 = frag_a(As, 40, wm * 32,      ln, hi);
      v16h a1 = frag_a(As, 40, wm * 32 + 16, ln, hi);
      v16h b0f = frag_b(Bs, 40, wn * 32,      ln, hi);
      v16h b1f = frag_b(Bs, 40, wn * 32 + 16, ln, hi);
      acc[0][0] = wmma16(a0, b0f, acc[0][0]);
      acc[0][1] = wmma16(a0, b1f, acc[0][1]);
      acc[1][0] = wmma16(a1, b0f, acc[1][0]);
      acc[1][1] = wmma16(a1, b1f, acc[1][1]);
      __syncthreads();
    }
  }
  #pragma unroll
  for (int i = 0; i < 2; ++i)
    #pragma unroll
    for (int j = 0; j < 2; ++j) {
      int cn = n0 + wn * 32 + j * 16 + ln;
      float bv = (cn < N) ? bias[cn] : 0.f;
      #pragma unroll
      for (int vi = 0; vi < 8; ++vi) {
        int rm = m0 + wm * 32 + i * 16 + vi + 8 * hi;
        if (rm < M && cn < N) {
          float val = acc[i][j][vi] + bv;
          if (relu) val = fmaxf(val, 0.f);
          if (Yf) Yf[(size_t)rm * N + cn] = val;
          else    Yh[(size_t)rm * N + cn] = (half_t)val;
        }
      }
    }
}

// ---------------------------------------------------------------------------
// Fused flash-style attention. qkv layout: [B*S, 1152] = [Q(384)|K(384)|V(384)],
// head h = cols h*192..h*192+191 of each third. Block = 128 threads,
// grid = (S/64, NHEADS, B). Each wave owns one 16-query tile; online softmax
// over 32-key chunks with key-padding bias. Q/K staged via async-to-LDS,
// V transposed through VGPRs, next chunk prefetched.
// Dynamic LDS layout (halfs): Qs 64*200 @0 | Ks 32*200 | Vt 192*40 | Ps 4*16*40
// ---------------------------------------------------------------------------
#define QS_H   (64 * 200)
#define KS_H   (32 * 200)
#define VT_H   (192 * 40)
#define PS_H   (4 * 16 * 40)
#define KS_OFF_B (QS_H * 2)
#define ATTN_LDS_BYTES ((QS_H + KS_H + VT_H + PS_H) * 2)

__global__ void __launch_bounds__(128) attn_kernel(
    const half_t* __restrict__ qkv, const float* __restrict__ kbias,
    half_t* __restrict__ out, int S)
{
  extern __shared__ half_t smem[];
  half_t* Qs = smem;
  half_t* Ks = Qs + QS_H;
  half_t* Vt = Ks + KS_H;
  half_t* Ps = Vt + VT_H;

  const int tid = threadIdx.x, lane = tid & 31, w = tid >> 5;
  const int hi = lane >> 4, ln = lane & 15;
  const int b = blockIdx.z, h = blockIdx.y;
  const int qb = blockIdx.x * 64;
  const size_t rs = 3 * EMB;
  const half_t* Qg = qkv + ((size_t)b * S + qb) * rs + (size_t)h * HD;
  const half_t* Kg = qkv + (size_t)b * S * rs + EMB     + (size_t)h * HD;
  const half_t* Vg = qkv + (size_t)b * S * rs + 2 * EMB + (size_t)h * HD;

  #pragma unroll
  for (int i = 0; i < 12; ++i) {            // 64*192/8 = 1536 chunks / 128 thr
    int c = tid + i * 128;
    int row = c / 24, kk0 = (c % 24) * 8;
    async_load_b128((uint32_t)((row * 200 + kk0) * 2), Qg + (size_t)row * rs + kk0);
  }

  v8f zz = {0.f,0.f,0.f,0.f,0.f,0.f,0.f,0.f};
  v8f o[12];
  #pragma unroll
  for (int nt = 0; nt < 12; ++nt) o[nt] = zz;
  float mrun[8], lrun[8];
  #pragma unroll
  for (int i = 0; i < 8; ++i) { mrun[i] = -1e30f; lrun[i] = 0.f; }
  const float sc = 0.0721687836487032f;     // 1/sqrt(192)

  for (int kc = 0; kc < S; kc += 32) {
    #pragma unroll
    for (int i = 0; i < 6; ++i) {           // 32*192/8 = 768 chunks
      int c = tid + i * 128;
      int row = c / 24, kk0 = (c % 24) * 8;
      async_load_b128((uint32_t)(KS_OFF_B + (row * 200 + kk0) * 2),
                      Kg + (size_t)(kc + row) * rs + kk0);
      union { uint4 q; half_t hh[8]; } u;
      u.q = *(const uint4*)(Vg + (size_t)(kc + row) * rs + kk0);
      #pragma unroll
      for (int q2 = 0; q2 < 8; ++q2) Vt[(kk0 + q2) * 40 + row] = u.hh[q2];
      if (kc + 32 < S) {                    // prefetch next key/value chunk
        __builtin_prefetch(Kg + (size_t)(kc + 32 + row) * rs + kk0, 0, 1);
        __builtin_prefetch(Vg + (size_t)(kc + 32 + row) * rs + kk0, 0, 1);
      }
    }
    wait_async0();
    __syncthreads();

    v8f s0 = zz, s1 = zz;
    #pragma unroll
    for (int t = 0; t < 6; ++t) {           // K = 192 = 6 x 32
      v16h qa  = frag_a(Qs + t * 32, 200, w * 16, ln, hi);
      v16h kb0 = frag_b(Ks + t * 32, 200, 0,  ln, hi);
      v16h kb1 = frag_b(Ks + t * 32, 200, 16, ln, hi);
      s0 = wmma16(qa, kb0, s0);
      s1 = wmma16(qa, kb1, s1);
    }
    float bb0 = kbias[(size_t)b * S + kc + ln];
    float bb1 = kbias[(size_t)b * S + kc + 16 + ln];
    #pragma unroll
    for (int i = 0; i < 8; ++i) {
      float x0 = s0[i] * sc + bb0;
      float x1 = s1[i] * sc + bb1;
      float rm = fmaxf(x0, x1);
      #pragma unroll
      for (int d = 1; d <= 8; d <<= 1) rm = fmaxf(rm, __shfl_xor(rm, d, 32));
      float mn = fmaxf(mrun[i], rm);
      float scl = __expf(mrun[i] - mn);
      mrun[i] = mn;
      float p0 = __expf(x0 - mn), p1 = __expf(x1 - mn);
      float rsum = p0 + p1;
      #pragma unroll
      for (int d = 1; d <= 8; d <<= 1) rsum += __shfl_xor(rsum, d, 32);
      lrun[i] = lrun[i] * scl + rsum;
      #pragma unroll
      for (int nt = 0; nt < 12; ++nt) o[nt][i] *= scl;
      s0[i] = p0; s1[i] = p1;
    }
    // C-layout -> A-layout via per-wave LDS round trip
    half_t* Pw = Ps + w * 640;
    #pragma unroll
    for (int i = 0; i < 8; ++i) {
      int row = i + 8 * hi;
      Pw[row * 40 + ln]      = (half_t)s0[i];
      Pw[row * 40 + 16 + ln] = (half_t)s1[i];
    }
    asm volatile("s_wait_dscnt 0x0" ::: "memory");
    v16h pa = frag_a(Pw, 40, 0, ln, hi);
    #pragma unroll
    for (int nt = 0; nt < 12; ++nt) {
      v16h vb = frag_b(Vt, 40, nt * 16, ln, hi);
      o[nt] = wmma16(pa, vb, o[nt]);
    }
    __syncthreads();
  }

  half_t* og = out + ((size_t)b * S + qb + w * 16) * EMB + (size_t)h * HD;
  #pragma unroll
  for (int nt = 0; nt < 12; ++nt)
    #pragma unroll
    for (int i = 0; i < 8; ++i) {
      float den = lrun[i];
      float val = (den > 0.f) ? o[nt][i] / den : 0.f;
      og[(size_t)(i + 8 * hi) * EMB + nt * 16 + ln] = (half_t)val;
    }
}

// ---------------------------------------------------------------------------
// out = [resid +] (LayerNorm(a) * g + b)  [ReLU].  One wave per row.
// ---------------------------------------------------------------------------
__global__ void __launch_bounds__(256) ln_kernel(
    const half_t* __restrict__ a, const half_t* __restrict__ resid,
    const float* __restrict__ g, const float* __restrict__ bt,
    half_t* __restrict__ out, int Mrows, int C, int relu)
{
  int w = threadIdx.x >> 5, lane = threadIdx.x & 31;
  int row = blockIdx.x * 8 + w;
  if (row >= Mrows) return;
  const half_t* ar = a + (size_t)row * C;
  float s = 0.f, s2 = 0.f;
  for (int i = lane; i < C; i += 32) { float x = (float)ar[i]; s += x; s2 += x * x; }
  for (int d = 16; d; d >>= 1) { s += __shfl_xor(s, d, 32); s2 += __shfl_xor(s2, d, 32); }
  float mu = s / C;
  float var = s2 / C - mu * mu;
  float rstd = rsqrtf(var + 1e-5f);
  const half_t* rr = resid ? resid + (size_t)row * C : nullptr;
  half_t* orow = out + (size_t)row * C;
  for (int i = lane; i < C; i += 32) {
    float y = ((float)ar[i] - mu) * rstd * g[i] + bt[i];
    if (rr) y += (float)rr[i];
    if (relu) y = fmaxf(y, 0.f);
    orow[i] = (half_t)y;
  }
}

__global__ void dp_lin_kernel(const half_t* __restrict__ h, const float* __restrict__ wv,
                              const float* __restrict__ bv, float* __restrict__ out,
                              int Mrows, int C)
{
  int w = threadIdx.x >> 5, lane = threadIdx.x & 31;
  int row = blockIdx.x * 8 + w;
  if (row >= Mrows) return;
  const half_t* hr = h + (size_t)row * C;
  float acc = 0.f;
  for (int i = lane; i < C; i += 32) acc += (float)hr[i] * wv[i];
  for (int d = 16; d; d >>= 1) acc += __shfl_xor(acc, d, 32);
  if (lane == 0) out[row] = fmaxf(acc + bv[0], 0.f);
}

__global__ void cast_kernel(const float* __restrict__ s, half_t* __restrict__ d, long n) {
  long i = (long)blockIdx.x * blockDim.x + threadIdx.x;
  if (i < n) d[i] = (half_t)s[i];
}

// w[l][n][c][t] (f32) -> o[l][t][n][c] (f16)
__global__ void prep_convw(const float* __restrict__ w, half_t* __restrict__ o,
                           int NL, int N, int C)
{
  long n = (long)blockIdx.x * blockDim.x + threadIdx.x;
  long tot = (long)NL * N * C * 3;
  if (n >= tot) return;
  int c = (int)(n % C); long r = n / C;
  int nn = (int)(r % N); r /= N;
  int t = (int)(r % 3);  int l = (int)(r / 3);
  o[n] = (half_t)w[(((long)l * N + nn) * C + c) * 3 + t];
}

__global__ void keybias_kernel(const int* __restrict__ ids, float* __restrict__ o, int n) {
  int i = blockIdx.x * blockDim.x + threadIdx.x;
  if (i < n) o[i] = (ids[i] == 0) ? -1e9f : 0.f;
}

__global__ void cumsum_kernel(const int* __restrict__ dur, int* __restrict__ cum) {
  int b = threadIdx.x;
  if (b < BB) {
    int s = 0;
    for (int l = 0; l < LL; ++l) { s += dur[b * LL + l]; cum[b * LL + l] = s; }
  }
}

__global__ void embed_kernel(const int* __restrict__ seq, const int* __restrict__ pos,
                             const float* __restrict__ tok, const float* __restrict__ pemb,
                             half_t* __restrict__ x)
{
  long n = (long)blockIdx.x * blockDim.x + threadIdx.x;
  if (n >= (long)BB * LL * EMB) return;
  int e = (int)(n % EMB);
  long r = n / EMB;
  x[n] = (half_t)(tok[(size_t)seq[r] * EMB + e] + pemb[(size_t)pos[r] * EMB + e]);
}

// length-regulate + decoder positional embedding
__global__ void lr_kernel(const half_t* __restrict__ henc, const int* __restrict__ cum,
                          const int* __restrict__ mel_pos, const float* __restrict__ dpos,
                          half_t* __restrict__ h)
{
  long n = (long)blockIdx.x * blockDim.x + threadIdx.x;
  if (n >= (long)BB * TT * EMB) return;
  int e = (int)(n % EMB);
  long r = n / EMB;
  int t = (int)(r % TT), b = (int)(r / TT);
  const int* cb = cum + b * LL;
  float v = 0.f;
  if (t < cb[LL - 1]) {
    int lo = 0, hh = LL;                // searchsorted(cum, t, 'right')
    while (lo < hh) { int mid = (lo + hh) >> 1; if (cb[mid] <= t) lo = mid + 1; else hh = mid; }
    if (lo > LL - 1) lo = LL - 1;
    v = (float)henc[((size_t)b * LL + lo) * EMB + e];
  }
  int p = mel_pos[(size_t)b * TT + t];
  v += dpos[(size_t)p * EMB + e];
  h[n] = (half_t)v;
}

__global__ void mask_kernel(half_t* __restrict__ h, const int* __restrict__ cum) {
  long n = (long)blockIdx.x * blockDim.x + threadIdx.x;
  if (n >= (long)BB * TT * EMB) return;
  long r = n / EMB;
  int t = (int)(r % TT), b = (int)(r / TT);
  if (t >= cum[b * LL + LL - 1]) h[n] = (half_t)0.f;
}

// ---------------------------------------------------------------------------
extern "C" void kernel_launch(void* const* d_in, const int* in_sizes, int n_in,
                              void* d_out, int out_size, void* d_ws, size_t ws_size,
                              hipStream_t stream)
{
  (void)in_sizes; (void)n_in; (void)out_size; (void)ws_size;
  // --- input mapping (setup_inputs dict order, nested dicts in insertion order) ---
  const int*   src_seq  = (const int*)d_in[0];
  const int*   src_pos  = (const int*)d_in[1];
  const int*   mel_pos  = (const int*)d_in[2];
  const int*   len_tgt  = (const int*)d_in[3];
  const float* tok_emb  = (const float*)d_in[4];
  const float* enc_pos  = (const float*)d_in[5];
  const float* dec_posE = (const float*)d_in[6];
  const float* ep[12]; for (int i = 0; i < 12; ++i) ep[i] = (const float*)d_in[7 + i];
  const float* dp[12]; for (int i = 0; i < 12; ++i) dp[i] = (const float*)d_in[19 + i];
  const float* dpp[10]; for (int i = 0; i < 10; ++i) dpp[i] = (const float*)d_in[31 + i];
  const float* mel_w = (const float*)d_in[41];
  const float* mel_b = (const float*)d_in[42];
  float* outf = (float*)d_out;

  const int Me = BB * LL;    // 4096 encoder rows
  const int Md = BB * TT;    // 32768 decoder rows

  // --- workspace bump allocator ---
  char* base = (char*)d_ws;
  size_t off = 0;
  auto alloc = [&](size_t bytes) -> char* {
    char* p = base + off; off += (bytes + 255) & ~(size_t)255; return p;
  };
  half_t* hbuf = (half_t*)alloc((size_t)Md * EMB * 2);
  half_t* buf2 = (half_t*)alloc((size_t)Md * EMB * 2);
  half_t* buf3 = (half_t*)alloc((size_t)Md * EMB * 2);
  half_t* big1 = (half_t*)alloc((size_t)Md * HID * 2);   // qkv / conv-hidden / DP
  half_t* henc = (half_t*)alloc((size_t)Me * EMB * 2);
  half_t* wqkv_e = (half_t*)alloc((size_t)NLAYERS * 3 * EMB * EMB * 2);
  half_t* wqkv_d = (half_t*)alloc((size_t)NLAYERS * 3 * EMB * EMB * 2);
  half_t* wout_e = (half_t*)alloc((size_t)NLAYERS * EMB * EMB * 2);
  half_t* wout_d = (half_t*)alloc((size_t)NLAYERS * EMB * EMB * 2);
  half_t* wc1_e  = (half_t*)alloc((size_t)NLAYERS * 3 * HID * EMB * 2);
  half_t* wc1_d  = (half_t*)alloc((size_t)NLAYERS * 3 * HID * EMB * 2);
  half_t* wc2_e  = (half_t*)alloc((size_t)NLAYERS * 3 * EMB * HID * 2);
  half_t* wc2_d  = (half_t*)alloc((size_t)NLAYERS * 3 * EMB * HID * 2);
  half_t* wdp1   = (half_t*)alloc((size_t)3 * HID * EMB * 2);
  half_t* wdp2   = (half_t*)alloc((size_t)3 * HID * HID * 2);
  half_t* wmel   = (half_t*)alloc((size_t)MELS * EMB * 2);
  float*  bias_e = (float*)alloc((size_t)Me * 4);
  float*  bias_d = (float*)alloc((size_t)Md * 4);
  int*    cum    = (int*)alloc((size_t)Me * 4);

  auto cg = [](long n) { return dim3((unsigned)((n + 255) / 256)); };

  // --- weight prep (f32 -> f16, conv weights re-laid to [tap][N][K]) ---
  long nq = (long)NLAYERS * 3 * EMB * EMB;
  cast_kernel<<<cg(nq), 256, 0, stream>>>(ep[0], wqkv_e, nq);
  cast_kernel<<<cg(nq), 256, 0, stream>>>(dp[0], wqkv_d, nq);
  long no = (long)NLAYERS * EMB * EMB;
  cast_kernel<<<cg(no), 256, 0, stream>>>(ep[2], wout_e, no);
  cast_kernel<<<cg(no), 256, 0, stream>>>(dp[2], wout_d, no);
  long nc1 = (long)NLAYERS * HID * EMB * 3;
  prep_convw<<<cg(nc1), 256, 0, stream>>>(ep[6], wc1_e, NLAYERS, HID, EMB);
  prep_convw<<<cg(nc1), 256, 0, stream>>>(dp[6], wc1_d, NLAYERS, HID, EMB);
  prep_convw<<<cg(nc1), 256, 0, stream>>>(ep[8], wc2_e, NLAYERS, EMB, HID);
  prep_convw<<<cg(nc1), 256, 0, stream>>>(dp[8], wc2_d, NLAYERS, EMB, HID);
  prep_convw<<<cg((long)HID * EMB * 3), 256, 0, stream>>>(dpp[0], wdp1, 1, HID, EMB);
  prep_convw<<<cg((long)HID * HID * 3), 256, 0, stream>>>(dpp[4], wdp2, 1, HID, HID);
  cast_kernel<<<cg((long)MELS * EMB), 256, 0, stream>>>(mel_w, wmel, (long)MELS * EMB);
  keybias_kernel<<<cg(Me), 256, 0, stream>>>(src_seq, bias_e, Me);
  keybias_kernel<<<cg(Md), 256, 0, stream>>>(mel_pos, bias_d, Md);
  cumsum_kernel<<<1, 32, 0, stream>>>(len_tgt, cum);

  // --- encoder ---
  embed_kernel<<<cg((long)Me * EMB), 256, 0, stream>>>(src_seq, src_pos, tok_emb, enc_pos, hbuf);
  for (int l = 0; l < NLAYERS; ++l) {
    gemm_conv_kernel<<<dim3(3 * EMB / 64, Me / 64), 128, GEMM_LDS_BYTES, stream>>>(
        hbuf, wqkv_e + (size_t)l * 3 * EMB * EMB, ep[1] + (size_t)l * 3 * EMB,
        big1, nullptr, Me, 3 * EMB, EMB, Me, 1, 0, 0);
    attn_kernel<<<dim3(LL / 64, NHEADS, BB), 128, ATTN_LDS_BYTES, stream>>>(
        big1, bias_e, buf2, LL);
    gemm_conv_kernel<<<dim3(EMB / 64, Me / 64), 128, GEMM_LDS_BYTES, stream>>>(
        buf2, wout_e + (size_t)l * EMB * EMB, ep[3] + (size_t)l * EMB,
        buf3, nullptr, Me, EMB, EMB, Me, 1, 0, 0);
    ln_kernel<<<dim3(Me / 8), 256, 0, stream>>>(buf3, hbuf, ep[4] + l * EMB, ep[5] + l * EMB,
                                                hbuf, Me, EMB, 0);
    gemm_conv_kernel<<<dim3(HID / 64, Me / 64), 128, GEMM_LDS_BYTES, stream>>>(
        hbuf, wc1_e + (size_t)l * 3 * HID * EMB, ep[7] + (size_t)l * HID,
        big1, nullptr, Me, HID, EMB, LL, 3, 1, 1);
    gemm_conv_kernel<<<dim3(EMB / 64, Me / 64), 128, GEMM_LDS_BYTES, stream>>>(
        big1, wc2_e + (size_t)l * 3 * EMB * HID, ep[9] + (size_t)l * EMB,
        buf2, nullptr, Me, EMB, HID, LL, 3, 1, 0);
    ln_kernel<<<dim3(Me / 8), 256, 0, stream>>>(buf2, hbuf, ep[10] + l * EMB, ep[11] + l * EMB,
                                                hbuf, Me, EMB, 0);
  }
  hipMemcpyAsync(henc, hbuf, (size_t)Me * EMB * 2, hipMemcpyDeviceToDevice, stream);

  // --- duration predictor (on encoder output) ---
  half_t* dpb = big1 + (size_t)Me * HID;
  gemm_conv_kernel<<<dim3(HID / 64, Me / 64), 128, GEMM_LDS_BYTES, stream>>>(
      henc, wdp1, dpp[1], big1, nullptr, Me, HID, EMB, LL, 3, 1, 0);
  ln_kernel<<<dim3(Me / 8), 256, 0, stream>>>(big1, nullptr, dpp[2], dpp[3], big1, Me, HID, 1);
  gemm_conv_kernel<<<dim3(HID / 64, Me / 64), 128, GEMM_LDS_BYTES, stream>>>(
      big1, wdp2, dpp[5], dpb, nullptr, Me, HID, HID, LL, 3, 1, 0);
  ln_kernel<<<dim3(Me / 8), 256, 0, stream>>>(dpb, nullptr, dpp[6], dpp[7], dpb, Me, HID, 1);
  float* dur_out = outf + (size_t)BB * TT * MELS;
  dp_lin_kernel<<<dim3(Me / 8), 256, 0, stream>>>(dpb, dpp[8], dpp[9], dur_out, Me, HID);

  // --- length regulator + decoder positional embedding ---
  lr_kernel<<<cg((long)Md * EMB), 256, 0, stream>>>(henc, cum, mel_pos, dec_posE, hbuf);

  // --- decoder ---
  for (int l = 0; l < NLAYERS; ++l) {
    gemm_conv_kernel<<<dim3(3 * EMB / 64, Md / 64), 128, GEMM_LDS_BYTES, stream>>>(
        hbuf, wqkv_d + (size_t)l * 3 * EMB * EMB, dp[1] + (size_t)l * 3 * EMB,
        big1, nullptr, Md, 3 * EMB, EMB, Md, 1, 0, 0);
    attn_kernel<<<dim3(TT / 64, NHEADS, BB), 128, ATTN_LDS_BYTES, stream>>>(
        big1, bias_d, buf2, TT);
    gemm_conv_kernel<<<dim3(EMB / 64, Md / 64), 128, GEMM_LDS_BYTES, stream>>>(
        buf2, wout_d + (size_t)l * EMB * EMB, dp[3] + (size_t)l * EMB,
        buf3, nullptr, Md, EMB, EMB, Md, 1, 0, 0);
    ln_kernel<<<dim3(Md / 8), 256, 0, stream>>>(buf3, hbuf, dp[4] + l * EMB, dp[5] + l * EMB,
                                                hbuf, Md, EMB, 0);
    gemm_conv_kernel<<<dim3(HID / 64, Md / 64), 128, GEMM_LDS_BYTES, stream>>>(
        hbuf, wc1_d + (size_t)l * 3 * HID * EMB, dp[7] + (size_t)l * HID,
        big1, nullptr, Md, HID, EMB, TT, 3, 1, 1);
    gemm_conv_kernel<<<dim3(EMB / 64, Md / 64), 128, GEMM_LDS_BYTES, stream>>>(
        big1, wc2_d + (size_t)l * 3 * EMB * HID, dp[9] + (size_t)l * EMB,
        buf2, nullptr, Md, EMB, HID, TT, 3, 1, 0);
    ln_kernel<<<dim3(Md / 8), 256, 0, stream>>>(buf2, hbuf, dp[10] + l * EMB, dp[11] + l * EMB,
                                                hbuf, Md, EMB, 0);
  }

  // --- mask padded frames, mel linear (f32 output straight into d_out) ---
  mask_kernel<<<cg((long)Md * EMB), 256, 0, stream>>>(hbuf, cum);
  gemm_conv_kernel<<<dim3((MELS + 63) / 64, Md / 64), 128, GEMM_LDS_BYTES, stream>>>(
      hbuf, wmel, mel_b, nullptr, outf, Md, MELS, EMB, Md, 1, 0, 0);
}
